// ERNN_K5_66228395704594
// MI455X (gfx1250) — matrix operand order repaired and checked by reading, not compiled, not run on hardware
//
#include <hip/hip_runtime.h>
#include <math.h>

// ---------------- types ----------------
typedef __bf16 bf16_t;
typedef __attribute__((ext_vector_type(16))) __bf16 v16bf;
typedef __attribute__((ext_vector_type(8)))  __bf16 v8bf;
typedef __attribute__((ext_vector_type(8)))  float  v8f;
typedef __attribute__((ext_vector_type(4)))  float  v4f;

constexpr int kD = 1024;   // feature dim
constexpr int kB = 64;     // batch
constexpr int kT = 200;    // time steps

// ---------------- helpers ----------------
__device__ __forceinline__ v8bf ld8bf(const bf16_t* p) {
  // 16-byte aligned vector load -> global_load_b128
  return __builtin_bit_cast(v8bf, *reinterpret_cast<const v4f*>(p));
}

__device__ __forceinline__ float eluf(float x) {
  return x > 0.0f ? x : expm1f(x);
}

// One wave accumulates a 16(M)x16(N) f32 tile over a K-window of KLEN:
//   A[16 x KLEN] * W[16 x KLEN]^T, both row-major with row stride kD (bf16).
// Fragment layouts per CDNA5 ISA 7.12.2 (wave32):
//   A 16x32: lane=(m,khalf); elems = K[khalf*8..+7] and K[16+khalf*8..+7]
//   B 32x16: lane=(n,khalf); elems = K[khalf*8..+7] and K[16+khalf*8..+7]
template <int KLEN>
__device__ __forceinline__ v8f wave_gemm_k(const bf16_t* __restrict__ A,
                                           const bf16_t* __restrict__ W,
                                           int lane) {
  const int idx  = lane & 15;
  const int half = lane >> 4;
  const bf16_t* ap = A + idx * kD + half * 8;    // + kb, + kb+16
  const bf16_t* bp = W + idx * kD + half * 8;    // + kb, + kb+16
  v8f acc = {0.f, 0.f, 0.f, 0.f, 0.f, 0.f, 0.f, 0.f};
#pragma unroll 4
  for (int kb = 0; kb < KLEN; kb += 32) {
    v8bf alo = ld8bf(ap + kb);
    v8bf ahi = ld8bf(ap + kb + 16);
    v8bf blo = ld8bf(bp + kb);
    v8bf bhi = ld8bf(bp + kb + 16);
    v16bf a = __builtin_shufflevector(alo, ahi, 0,1,2,3,4,5,6,7,8,9,10,11,12,13,14,15);
    v16bf b = __builtin_shufflevector(blo, bhi, 0,1,2,3,4,5,6,7,8,9,10,11,12,13,14,15);
    acc = __builtin_amdgcn_wmma_f32_16x16x32_bf16(false, a, false, b,
                                                  (short)0, acc, false, false);
  }
  return acc;
}

// C/D tile layout (32-bit, 16x16): lane l holds N = l&15; element r holds
// M = r + 8*(l>>4).

// Critical-path stage tile: block = 512 threads = 16 waves on ONE N-tile.
// wave = (mTile in 0..3, ksplit in 0..3); each wave does K/4 = 256 (8 WMMAs),
// partials reduced through LDS. Returns true iff this wave (ksplit==0) holds
// the final sum and must run the epilogue.
__device__ __forceinline__ bool stage_tile(const bf16_t* __restrict__ A,
                                           const bf16_t* __restrict__ Wn0,
                                           v8f& acc, int& m0, int& lane) {
  __shared__ float red[12][8][32];       // 12 KB, [slot][reg][lane] -> no conflicts
  const int tid  = threadIdx.x;
  lane = tid & 31;
  const int wave  = tid >> 5;
  const int mTile = wave & 3;
  const int ks    = wave >> 2;
  m0 = mTile * 16;
  acc = wave_gemm_k<256>(A + m0 * kD + ks * 256, Wn0 + ks * 256, lane);
  if (ks != 0) {
    float (*slab)[32] = red[(ks - 1) * 4 + mTile];
#pragma unroll
    for (int r = 0; r < 8; ++r) slab[r][lane] = acc[r];
  }
  __syncthreads();
  if (ks != 0) return false;
#pragma unroll
  for (int s = 0; s < 3; ++s) {
#pragma unroll
    for (int r = 0; r < 8; ++r) acc[r] += red[s * 4 + mTile][r][lane];
  }
  return true;
}

// ---------------- prep kernels ----------------
__global__ void k_cvt_bf16(const float* __restrict__ s, bf16_t* __restrict__ d, int n) {
  int i = blockIdx.x * blockDim.x + threadIdx.x;
  if (i < n) d[i] = (bf16_t)s[i];
}

// XT[t][b][d] (bf16) = x[b][d][t] (f32), LDS-tiled 32x32 transpose per b-slice
__global__ void k_transpose_x(const float* __restrict__ x, bf16_t* __restrict__ XT) {
  __shared__ float tile[32][33];
  const int b  = blockIdx.z;
  const int d0 = blockIdx.x * 32;
  const int t0 = blockIdx.y * 32;
  const int tx = threadIdx.x, ty = threadIdx.y;
#pragma unroll
  for (int i = 0; i < 4; ++i) {
    int d = d0 + ty + i * 8;
    int t = t0 + tx;
    tile[ty + i * 8][tx] =
        (t < kT) ? x[(size_t)b * kD * kT + (size_t)d * kT + t] : 0.0f;
  }
  __syncthreads();
#pragma unroll
  for (int i = 0; i < 4; ++i) {
    int t = t0 + ty + i * 8;
    int d = d0 + tx;
    if (t < kT)
      XT[((size_t)t * kB + b) * kD + d] = (bf16_t)tile[tx][ty + i * 8];
  }
}

__global__ void k_init_state(float* __restrict__ h, float* __restrict__ h0,
                             bf16_t* __restrict__ Hs, int n) {
  int i = blockIdx.x * blockDim.x + threadIdx.x;
  if (i < n) { h[i] = 0.0f; h0[i] = 0.0f; Hs[i] = (bf16_t)0.0f; }
}

// ---------------- batched GEMMs (off the critical path) ----------------
// WX[t] = XT[t] @ W_in^T + b_in   (f32 out), grid (64, T), 4 waves = 4 M-tiles
__global__ void k_wx(const bf16_t* __restrict__ XT, const bf16_t* __restrict__ Wb,
                     const float* __restrict__ bias, float* __restrict__ WX) {
  const int lane = threadIdx.x & 31, wave = threadIdx.x >> 5;
  const int m0 = wave * 16, n0 = blockIdx.x * 16, t = blockIdx.y;
  const bf16_t* A = XT + (size_t)t * kB * kD;
  v8f acc = wave_gemm_k<kD>(A + m0 * kD, Wb + n0 * kD, lane);
  const int n = n0 + (lane & 15), half = lane >> 4;
  const float bn = bias[n];
  float* Ot = WX + (size_t)t * kB * kD;
#pragma unroll
  for (int r = 0; r < 8; ++r) {
    int m = m0 + r + 8 * half;
    Ot[m * kD + n] = acc[r] + bn;
  }
}

// out[b][d][t] = Hall[t][b] @ W_out^T + b_out
__global__ void k_y(const bf16_t* __restrict__ Hall, const bf16_t* __restrict__ Wb,
                    const float* __restrict__ bias, float* __restrict__ out) {
  const int lane = threadIdx.x & 31, wave = threadIdx.x >> 5;
  const int m0 = wave * 16, n0 = blockIdx.x * 16, t = blockIdx.y;
  const bf16_t* A = Hall + (size_t)t * kB * kD;
  v8f acc = wave_gemm_k<kD>(A + m0 * kD, Wb + n0 * kD, lane);
  const int n = n0 + (lane & 15), half = lane >> 4;
  const float bn = bias[n];
#pragma unroll
  for (int r = 0; r < 8; ++r) {
    int m = m0 + r + 8 * half;   // m == batch index b
    out[(size_t)m * (kD * kT) + (size_t)n * kT + t] = acc[r] + bn;
  }
}

// ---------------- recurrence stage kernels (grid 64, block 512) ----------------
// S1: F1 = elu( Hs @ W_hid^T + b_hid + WX[t] ),  Hs = bf16(h + h0)
__global__ void k_s1(const bf16_t* __restrict__ Hs, const bf16_t* __restrict__ Wb,
                     const float* __restrict__ bias, const float* __restrict__ WXt,
                     bf16_t* __restrict__ F1) {
  const int n0 = blockIdx.x * 16;
  v8f acc; int m0, lane;
  if (!stage_tile(Hs, Wb + n0 * kD, acc, m0, lane)) return;
  const int n = n0 + (lane & 15), half = lane >> 4;
  const float bn = bias[n];
#pragma unroll
  for (int r = 0; r < 8; ++r) {
    int m = m0 + r + 8 * half;
    int i = m * kD + n;
    F1[i] = (bf16_t)eluf(acc[r] + bn + WXt[i]);
  }
}

// S2: F2 = elu( F1 @ W2^T + b2 )
__global__ void k_s2(const bf16_t* __restrict__ F1, const bf16_t* __restrict__ Wb,
                     const float* __restrict__ bias, bf16_t* __restrict__ F2) {
  const int n0 = blockIdx.x * 16;
  v8f acc; int m0, lane;
  if (!stage_tile(F1, Wb + n0 * kD, acc, m0, lane)) return;
  const int n = n0 + (lane & 15), half = lane >> 4;
  const float bn = bias[n];
#pragma unroll
  for (int r = 0; r < 8; ++r) {
    int m = m0 + r + 8 * half;
    F2[m * kD + n] = (bf16_t)eluf(acc[r] + bn);
  }
}

// S3: Fn = elu( F2 @ W3^T + b3 );  h += eta*(Fn - (h + h0))
//     writes next A-operand Hs = bf16(h_new + h0); on last iter also commits
//     h0 <- h_new, Hall[t] <- bf16(h_new), Hs <- bf16(2*h_new)
__global__ void k_s3(const bf16_t* __restrict__ F2, const bf16_t* __restrict__ Wb,
                     const float* __restrict__ bias, const float* __restrict__ etas,
                     int kIter, float* __restrict__ h, float* __restrict__ h0,
                     bf16_t* __restrict__ Hs, bf16_t* __restrict__ Hallt, int last) {
  const int n0 = blockIdx.x * 16;
  v8f acc; int m0, lane;
  if (!stage_tile(F2, Wb + n0 * kD, acc, m0, lane)) return;
  const int n = n0 + (lane & 15), half = lane >> 4;
  const float bn  = bias[n];
  const float eta = etas[kIter];
#pragma unroll
  for (int r = 0; r < 8; ++r) {
    int m = m0 + r + 8 * half;
    int i = m * kD + n;
    float fn  = eluf(acc[r] + bn);
    float hv  = h[i];
    float h0v = h0[i];
    float hn  = hv + eta * (fn - hv - h0v);
    h[i] = hn;
    if (last) {
      h0[i]    = hn;                  // becomes next step's h0
      Hs[i]    = (bf16_t)(2.0f * hn); // next step iter1: h + h0 = 2*h
      Hallt[i] = (bf16_t)hn;          // saved state for output GEMM
    } else {
      Hs[i] = (bf16_t)(hn + h0v);
    }
  }
}

// ---------------- host launcher ----------------
extern "C" void kernel_launch(void* const* d_in, const int* in_sizes, int n_in,
                              void* d_out, int out_size, void* d_ws, size_t ws_size,
                              hipStream_t stream) {
  (void)in_sizes; (void)n_in; (void)out_size; (void)ws_size;

  const float* x     = (const float*)d_in[0];
  const float* W_in  = (const float*)d_in[1];
  const float* b_in  = (const float*)d_in[2];
  const float* W_hid = (const float*)d_in[3];
  const float* b_hid = (const float*)d_in[4];
  const float* W2    = (const float*)d_in[5];
  const float* b2    = (const float*)d_in[6];
  const float* W3    = (const float*)d_in[7];
  const float* b3    = (const float*)d_in[8];
  const float* W_out = (const float*)d_in[9];
  const float* b_out = (const float*)d_in[10];
  const float* etas  = (const float*)d_in[11];
  float* out = (float*)d_out;

  uint8_t* ws = (uint8_t*)d_ws;
  size_t off = 0;
  auto carve = [&](size_t bytes) -> void* {
    void* p = ws + off;
    off += (bytes + 255) & ~(size_t)255;
    return p;
  };
  const size_t WElems  = (size_t)kD * kD;          // 1M
  const size_t SElems  = (size_t)kB * kD;          // 64K (one [B,D] slab)
  const size_t TBElems = (size_t)kT * kB * kD;     // 12.8M

  bf16_t* Winb  = (bf16_t*)carve(WElems * 2);
  bf16_t* Whidb = (bf16_t*)carve(WElems * 2);
  bf16_t* W2b   = (bf16_t*)carve(WElems * 2);
  bf16_t* W3b   = (bf16_t*)carve(WElems * 2);
  bf16_t* Woutb = (bf16_t*)carve(WElems * 2);
  bf16_t* XT    = (bf16_t*)carve(TBElems * 2);     // x transposed, bf16
  float*  WX    = (float*) carve(TBElems * 4);     // x_t @ W_in^T + b_in
  bf16_t* Hall  = (bf16_t*)carve(TBElems * 2);     // h_t after each step
  float*  h     = (float*) carve(SElems * 4);
  float*  h0    = (float*) carve(SElems * 4);
  bf16_t* Hs    = (bf16_t*)carve(SElems * 2);      // bf16(h + h0)
  bf16_t* F1    = (bf16_t*)carve(SElems * 2);
  bf16_t* F2    = (bf16_t*)carve(SElems * 2);

  // 1) convert weights to bf16 (stay resident in L2 across the whole run)
  {
    int n = (int)WElems, g = (n + 255) / 256;
    k_cvt_bf16<<<g, 256, 0, stream>>>(W_in,  Winb,  n);
    k_cvt_bf16<<<g, 256, 0, stream>>>(W_hid, Whidb, n);
    k_cvt_bf16<<<g, 256, 0, stream>>>(W2,    W2b,   n);
    k_cvt_bf16<<<g, 256, 0, stream>>>(W3,    W3b,   n);
    k_cvt_bf16<<<g, 256, 0, stream>>>(W_out, Woutb, n);
  }

  // 2) transpose x [B,D,T] -> XT [T,B,D] bf16
  k_transpose_x<<<dim3(kD / 32, (kT + 31) / 32, kB), dim3(32, 8), 0, stream>>>(x, XT);

  // 3) zero recurrent state
  k_init_state<<<((int)SElems + 255) / 256, 256, 0, stream>>>(h, h0, Hs, (int)SElems);

  // 4) batched input GEMM: WX[t] = XT[t] @ W_in^T + b_in
  k_wx<<<dim3(kD / 16, kT), 128, 0, stream>>>(XT, Winb, b_in, WX);

  // 5) recurrence: 200 steps x 5 iterations x 3 dependent GEMM stages
  //    (grid 64 N-tiles, 16 waves/block: 4 M-tiles x 4-way K-split + LDS reduce)
  for (int t = 0; t < kT; ++t) {
    const float* WXt = WX + (size_t)t * SElems;
    bf16_t* Hallt = Hall + (size_t)t * SElems;
    for (int k = 0; k < 5; ++k) {
      k_s1<<<kD / 16, 512, 0, stream>>>(Hs, Whidb, b_hid, WXt, F1);
      k_s2<<<kD / 16, 512, 0, stream>>>(F1, W2b, b2, F2);
      k_s3<<<kD / 16, 512, 0, stream>>>(F2, W3b, b3, etas, k, h, h0, Hs, Hallt,
                                        (k == 4) ? 1 : 0);
    }
  }

  // 6) batched output GEMM with strided [B,D,T] epilogue
  k_y<<<dim3(kD / 16, kT), 128, 0, stream>>>(Hall, Woutb, b_out, out);
}
